// RNNTSP_57294863729275
// MI455X (gfx1250) — compile-verified, matrix operand order checked by reading, not run on hardware
//
#include <hip/hip_runtime.h>
#include <stdint.h>
#include <stddef.h>

// Problem sizes (fixed by the reference)
#define Bn 1024
#define Sn 100
#define Hn 256
#define Gn 1024  // 4*H

typedef __attribute__((ext_vector_type(16))) __bf16 v16bf;
typedef __attribute__((ext_vector_type(8)))  __bf16 v8bf;
typedef __attribute__((ext_vector_type(8)))  float  v8f;
typedef __attribute__((ext_vector_type(4)))  unsigned int v4u;
typedef __attribute__((ext_vector_type(8)))  int v8i;
typedef __attribute__((ext_vector_type(4)))  int v4i;

#if __has_builtin(__builtin_amdgcn_tensor_load_to_lds)
#define HAVE_TDM 1
#else
#define HAVE_TDM 0
#endif

static __device__ __forceinline__ float sigm(float x) { return 1.0f / (1.0f + __expf(-x)); }

static __device__ __forceinline__ float urand01(uint32_t x) {
  x ^= x >> 16; x *= 0x7feb352dU; x ^= x >> 15; x *= 0x846ca68bU; x ^= x >> 16;
  return ((x >> 8) + 1) * (1.0f / 16777217.0f);
}

// ---------------------------------------------------------------------------
// TDM 2D tile load: global (row-major, 4-byte units) -> LDS, rows padded
// 64B -> 96B (pad_interval=16 DWORDs, pad_amount=8 DWORDs) to match LDT=48.
// D# layout per CDNA5 ISA ch.8: group0 = {flags, lds_addr, global_addr, type=2},
// group1 = {data_size/pad, tensor dims, tile dims, dim0 stride}.
// This toolchain exposes the 6-arg builtin form:
//   (uint32x4 g0, int32x8 g1, int32x4 g2, int32x4 g3, int32x8 extra, i32 cpol)
// ---------------------------------------------------------------------------
#if HAVE_TDM
static __device__ __forceinline__ void tdm_load_2d(const void* gsrc, uint32_t lds_off,
                                                   uint32_t row_elems4, uint32_t nrows,
                                                   uint32_t stride_elems4,
                                                   uint32_t tile_c4, uint32_t tile_r)
{
  const uint64_t ga = (uint64_t)(size_t)gsrc;
  v4u g0;
  g0.x = 1u;                                                  // count=1, user descriptor
  g0.y = lds_off;                                             // LDS byte address
  g0.z = (uint32_t)ga;                                        // global_addr[31:0]
  g0.w = (uint32_t)((ga >> 32) & 0x01ffffffu) | (2u << 30);   // global_addr[56:32] | type=2
  v8i g1;
  // data_size=2 (4B) <<16 | pad_enable<<20 | pad_interval=3 (16 DW)<<22 | pad_amount=7 (8 DW)<<25
  g1[0] = (int)((2u << 16) | (1u << 20) | (3u << 22) | (7u << 25));
  g1[1] = (int)((row_elems4 & 0xffffu) << 16);                            // tensor_dim0[15:0]
  g1[2] = (int)(((row_elems4 >> 16) & 0xffffu) | ((nrows & 0xffffu) << 16));   // dim0 hi | dim1 lo
  g1[3] = (int)(((nrows >> 16) & 0xffffu) | ((tile_c4 & 0xffffu) << 16));      // dim1 hi | tile_dim0
  g1[4] = (int)(tile_r & 0xffffu);                            // tile_dim1 (tile_dim2=0)
  g1[5] = (int)stride_elems4;                                 // tensor_dim0_stride[31:0]
  g1[6] = 0;                                                  // stride[47:32]=0, dim1_stride lo=0
  g1[7] = 0;
  v4i gz4 = {0, 0, 0, 0};
  v8i gz8 = {0, 0, 0, 0, 0, 0, 0, 0};
  __builtin_amdgcn_tensor_load_to_lds(g0, g1, gz4, gz4, gz8, 0);
}
#endif

// ---------------------------------------------------------------------------
// Tiled bf16 WMMA GEMM:  C[M,N] = A1[M,K1]*W1[N,K1]^T + A2[M,K2]*W2[N,K2]^T + bias[N]
// Block: 256 threads = 8 waves, tile 32(M) x 64(N), K chunked by 32.
// TDM double-buffered LDS pipeline (fallback: synchronous cooperative loads).
// ---------------------------------------------------------------------------
#define TM 32
#define TN 64
#define TK 32
#define LDT 48  // padded LDS row stride in bf16 elems (96B, 16B-aligned rows)

__global__ __launch_bounds__(256)
void ptrnet_gemm_bf16_wmma(const __bf16* __restrict__ A1, int lda1,
                           const __bf16* __restrict__ W1,
                           const __bf16* __restrict__ A2, int lda2,
                           const __bf16* __restrict__ W2,
                           const float* __restrict__ bias,
                           float* __restrict__ Cf, __bf16* __restrict__ Cbf,
                           int M, int N, int K1, int K2)
{
  __shared__ __bf16 As[2][TM * LDT];
  __shared__ __bf16 Bs[2][TN * LDT];

  const int tid  = threadIdx.x;
  const int lane = tid & 31;
  const int wave = tid >> 5;
  const int nTilesN = N / TN;
  const int bm = (blockIdx.x / nTilesN) * TM;
  const int bn = (blockIdx.x % nTilesN) * TN;
  const int wr = wave >> 2;            // 0..1 (M direction)
  const int wc = wave & 3;             // 0..3 (N direction)
  const int kh = (lane < 16) ? 0 : 8;  // per-lane K half per 16-bit A/B layout

  const int n1  = K1 / TK;
  const int n2  = (A2 != nullptr && K2 > 0) ? (K2 / TK) : 0;
  const int nch = n1 + n2;

  auto resolve = [&](int ch, const __bf16*& A, const __bf16*& W, int& lda, int& K, int& k0) {
    if (ch < n1) { A = A1; W = W1; lda = lda1; K = K1; k0 = ch * TK; }
    else         { A = A2; W = W2; lda = lda2; K = K2; k0 = (ch - n1) * TK; }
  };

  v8f acc = {};

#if HAVE_TDM
  auto issueChunk = [&](int ch, int buf) {
    const __bf16* A; const __bf16* W; int lda, K, k0;
    resolve(ch, A, W, lda, K, k0);
    tdm_load_2d(A + (size_t)bm * (size_t)lda + (size_t)k0,
                (uint32_t)(size_t)&As[buf][0],
                (uint32_t)(K >> 1), (uint32_t)M, (uint32_t)(lda >> 1),
                TK / 2, TM);
    tdm_load_2d(W + (size_t)bn * (size_t)K + (size_t)k0,
                (uint32_t)(size_t)&Bs[buf][0],
                (uint32_t)(K >> 1), (uint32_t)N, (uint32_t)(K >> 1),
                TK / 2, TN);
  };
  if (tid == 0) issueChunk(0, 0);
#endif

  for (int ch = 0; ch < nch; ++ch) {
#if HAVE_TDM
    const int buf = ch & 1;
    if (tid == 0) {
      if (ch + 1 < nch) {
        issueChunk(ch + 1, buf ^ 1);
        __builtin_amdgcn_s_wait_tensorcnt(2);  // chunk ch landed (per-wave in-order)
      } else {
        __builtin_amdgcn_s_wait_tensorcnt(0);
      }
    }
#else
    const int buf = 0;
    const __bf16* A; const __bf16* W; int lda, K, k0;
    resolve(ch, A, W, lda, K, k0);
    // A tile 32x32 bf16: threads 0..127 load 8 elems (16B) each
    if (tid < 128) {
      const int r = tid >> 2;
      const int c = (tid & 3) * 8;
      const __bf16* src = A + (size_t)(bm + r) * (size_t)lda + (size_t)(k0 + c);
      *(uint4*)(&As[buf][r * LDT + c]) = *(const uint4*)src;
      __builtin_prefetch(src + TK, 0, 1);
    }
    // B tile 64x32 bf16: all 256 threads load 8 elems (16B) each
    {
      const int r = tid >> 2;
      const int c = (tid & 3) * 8;
      const __bf16* src = W + (size_t)(bn + r) * (size_t)K + (size_t)(k0 + c);
      *(uint4*)(&Bs[buf][r * LDT + c]) = *(const uint4*)src;
      __builtin_prefetch(src + TK, 0, 1);
    }
#endif
    __syncthreads();

    // Fragment loads from LDS (ds_load_b128), per ISA 16-bit A/B VGPR layout:
    // lane<16 holds K {0..7,16..23}, lane>=16 holds K {8..15,24..31}
    const int rl = wr * 16 + (lane & 15);
    const int cl = wc * 16 + (lane & 15);
    v8bf alo = *(const v8bf*)(&As[buf][rl * LDT + kh]);
    v8bf ahi = *(const v8bf*)(&As[buf][rl * LDT + kh + 16]);
    v8bf blo = *(const v8bf*)(&Bs[buf][cl * LDT + kh]);
    v8bf bhi = *(const v8bf*)(&Bs[buf][cl * LDT + kh + 16]);
    v16bf a = __builtin_shufflevector(alo, ahi, 0,1,2,3,4,5,6,7,8,9,10,11,12,13,14,15);
    v16bf b = __builtin_shufflevector(blo, bhi, 0,1,2,3,4,5,6,7,8,9,10,11,12,13,14,15);

    acc = __builtin_amdgcn_wmma_f32_16x16x32_bf16(
        /*neg_a=*/false, a, /*neg_b=*/false, b,
        /*c_mod=*/(short)0, acc, /*reuse_a=*/false, /*reuse_b=*/false);

    __syncthreads();
  }

  // Store: C/D layout — VGPR i: lane<16 -> M=i, lane>=16 -> M=8+i; N = lane&15
  const int cc = bn + wc * 16 + (lane & 15);
  const float bval = bias ? bias[cc] : 0.0f;
  const int rbase = bm + wr * 16 + ((lane < 16) ? 0 : 8);
#pragma unroll
  for (int i = 0; i < 8; ++i) {
    const float vv = acc[i] + bval;
    const size_t idx = (size_t)(rbase + i) * (size_t)N + (size_t)cc;
    if (Cbf) Cbf[idx] = (__bf16)vv;
    else     Cf[idx]  = vv;
  }
}

// ---------------------------------------------------------------------------
// Elementwise / small kernels
// ---------------------------------------------------------------------------
__global__ void ptrnet_cvt_f2bf(const float* __restrict__ s, __bf16* __restrict__ d, int n) {
  int i = blockIdx.x * blockDim.x + threadIdx.x;
  if (i < n) d[i] = (__bf16)s[i];
}

__global__ void ptrnet_addvec(const float* __restrict__ a, const float* __restrict__ b,
                              float* __restrict__ d, int n) {
  int i = blockIdx.x * blockDim.x + threadIdx.x;
  if (i < n) d[i] = a[i] + b[i];
}

__global__ void ptrnet_fillf(float* __restrict__ d, int n) {
  int i = blockIdx.x * blockDim.x + threadIdx.x;
  if (i < n) d[i] = 0.0f;
}

__global__ void ptrnet_fillbf(__bf16* __restrict__ d, int n) {
  int i = blockIdx.x * blockDim.x + threadIdx.x;
  if (i < n) d[i] = (__bf16)0.0f;
}

__global__ void ptrnet_filli(int* __restrict__ d, int n) {
  int i = blockIdx.x * blockDim.x + threadIdx.x;
  if (i < n) d[i] = 0;
}

// embedded[b,s,:] = inputs[b,s,:2] @ emb_w^T + emb_b  (stored bf16)
__global__ void ptrnet_embed(const float* __restrict__ in2, const float* __restrict__ emb_w,
                             const float* __restrict__ emb_b, __bf16* __restrict__ out) {
  int i = blockIdx.x * blockDim.x + threadIdx.x;  // over B*S*H
  if (i >= Bn * Sn * Hn) return;
  int bs = i / Hn;
  int j  = i - bs * Hn;
  float e = in2[bs * 2 + 0] * emb_w[j * 2 + 0] + in2[bs * 2 + 1] * emb_w[j * 2 + 1] + emb_b[j];
  out[i] = (__bf16)e;
}

// dec_in[b,:] = start[:]
__global__ void ptrnet_start_bcast(const float* __restrict__ startv, __bf16* __restrict__ d) {
  int i = blockIdx.x * blockDim.x + threadIdx.x;  // over B*H
  if (i < Bn * Hn) d[i] = (__bf16)startv[i % Hn];
}

// LSTM cell: gates [B,4H] (i,f,g,o), updates c/h (f32), h_bf (bf16), optional
// enc_store base = enc_out_bf + t*H, indexed b*(S*H)+j.
__global__ void ptrnet_lstm_cell(const float* __restrict__ gates,
                                 float* __restrict__ h, float* __restrict__ c,
                                 __bf16* __restrict__ h_bf, __bf16* __restrict__ enc_store) {
  int i = blockIdx.x * blockDim.x + threadIdx.x;  // over B*H
  if (i >= Bn * Hn) return;
  int b = i / Hn;
  int j = i - b * Hn;
  const float* g = gates + (size_t)b * Gn;
  float gi = g[j], gf = g[Hn + j], gg = g[2 * Hn + j], go = g[3 * Hn + j];
  float cn = sigm(gf) * c[i] + sigm(gi) * tanhf(gg);
  float hn = sigm(go) * tanhf(cn);
  c[i] = cn; h[i] = hn;
  h_bf[i] = (__bf16)hn;
  if (enc_store) enc_store[(size_t)b * (Sn * Hn) + j] = (__bf16)hn;
}

// Attention over S for one batch element per block (256 threads = 8 waves).
// u = tanh(ref[b,s,:] + qproj[b,:]); raw = u.v + v_b; masked -> NEG; lg = 10*tanh(raw)
// sample==0: write softmax probs.  sample==1: log-softmax + Gumbel-argmax choice.
__global__ __launch_bounds__(256)
void ptrnet_attn(const __bf16* __restrict__ ref, const float* __restrict__ qproj,
                 const float* __restrict__ v_w, const float* __restrict__ v_b,
                 int* __restrict__ mask, float* __restrict__ probs,
                 float* __restrict__ out_lp, float* __restrict__ out_ch,
                 int* __restrict__ chosen_i, int step, int sample)
{
  __shared__ float qp[Hn];
  __shared__ float vv[Hn];
  __shared__ float lg[128];

  const int b = blockIdx.x;
  const int tid = threadIdx.x;
  const int lane = tid & 31;
  const int wv = tid >> 5;

  qp[tid] = qproj[(size_t)b * Hn + tid];
  vv[tid] = v_w[tid];
  __syncthreads();

  for (int s = wv; s < Sn; s += 8) {
    const __bf16* r = ref + ((size_t)b * Sn + s) * Hn;
    float p = 0.0f;
#pragma unroll
    for (int jj = 0; jj < Hn / 32; ++jj) {
      int j = lane + jj * 32;
      float u = tanhf((float)r[j] + qp[j]);
      p += u * vv[j];
    }
#pragma unroll
    for (int off = 16; off; off >>= 1) p += __shfl_xor(p, off, 32);
    if (lane == 0) {
      float raw = p + v_b[0];
      if (mask[b * Sn + s]) raw = -100000.0f;
      lg[s] = 10.0f * tanhf(raw);
    }
  }
  __syncthreads();

  // block-redundant softmax stats over S=100 (cheap, all from LDS)
  float mx = -1e30f;
  for (int s = 0; s < Sn; ++s) mx = fmaxf(mx, lg[s]);
  float sum = 0.0f;
  for (int s = 0; s < Sn; ++s) sum += __expf(lg[s] - mx);

  if (sample == 0) {
    for (int s = tid; s < Sn; s += 256)
      probs[b * Sn + s] = __expf(lg[s] - mx) / sum;
  } else if (tid == 0) {
    float logZ = mx + __logf(sum);
    float best = -1e30f; int bi = 0;
    for (int s = 0; s < Sn; ++s) {
      uint32_t seed = (uint32_t)(b * 1000003) ^ (uint32_t)(step * 7919) ^ (uint32_t)(s * 104729) ^ 0x9e3779b9u;
      float u = urand01(seed);
      float g = -__logf(-__logf(u));
      float sc = lg[s] + g;
      if (sc > best) { best = sc; bi = s; }
    }
    out_lp[b * Sn + step] = lg[bi] - logZ;
    out_ch[b * Sn + step] = (float)bi;
    chosen_i[b] = bi;
    mask[b * Sn + bi] = 1;
  }
}

// q[b,j] = sum_s probs[b,s] * enc_out[b,s,j]  (bf16 out for next GEMM)
__global__ void ptrnet_glimpse(const float* __restrict__ probs,
                               const __bf16* __restrict__ enc_out,
                               __bf16* __restrict__ q_bf) {
  int i = blockIdx.x * blockDim.x + threadIdx.x;  // over B*H
  if (i >= Bn * Hn) return;
  int b = i / Hn;
  int j = i - b * Hn;
  const float* p = probs + b * Sn;
  const __bf16* e = enc_out + (size_t)b * Sn * Hn + j;
  float acc = 0.0f;
  for (int s = 0; s < Sn; ++s) acc += p[s] * (float)e[(size_t)s * Hn];
  q_bf[i] = (__bf16)acc;
}

// dec_in[b,:] = embedded[b, chosen[b], :]
__global__ void ptrnet_gather(const __bf16* __restrict__ emb,
                              const int* __restrict__ chosen_i,
                              __bf16* __restrict__ dec_in) {
  int b = blockIdx.x;
  int j = threadIdx.x;
  dec_in[(size_t)b * Hn + j] = emb[((size_t)b * Sn + chosen_i[b]) * Hn + j];
}

// ---------------------------------------------------------------------------
// Host orchestration
// ---------------------------------------------------------------------------
extern "C" void kernel_launch(void* const* d_in, const int* in_sizes, int n_in,
                              void* d_out, int out_size, void* d_ws, size_t ws_size,
                              hipStream_t stream)
{
  (void)in_sizes; (void)n_in; (void)out_size; (void)ws_size;

  const float* inputs     = (const float*)d_in[0];
  const float* emb_w      = (const float*)d_in[1];
  const float* emb_b      = (const float*)d_in[2];
  const float* enc_wih    = (const float*)d_in[3];
  const float* enc_whh    = (const float*)d_in[4];
  const float* enc_bih    = (const float*)d_in[5];
  const float* enc_bhh    = (const float*)d_in[6];
  const float* dec_wih    = (const float*)d_in[7];
  const float* dec_whh    = (const float*)d_in[8];
  const float* dec_bih    = (const float*)d_in[9];
  const float* dec_bhh    = (const float*)d_in[10];
  const float* ptr_wq_w   = (const float*)d_in[11];
  const float* ptr_wq_b   = (const float*)d_in[12];
  const float* ptr_wref_w = (const float*)d_in[13];
  const float* ptr_wref_b = (const float*)d_in[14];
  const float* ptr_v_w    = (const float*)d_in[15];
  const float* ptr_v_b    = (const float*)d_in[16];
  const float* glm_wq_w   = (const float*)d_in[17];
  const float* glm_wq_b   = (const float*)d_in[18];
  const float* glm_wref_w = (const float*)d_in[19];
  const float* glm_wref_b = (const float*)d_in[20];
  const float* glm_v_w    = (const float*)d_in[21];
  const float* glm_v_b    = (const float*)d_in[22];
  const float* startv     = (const float*)d_in[23];

  // Bump allocator over workspace (~220 MB total)
  size_t off = 0;
  auto alloc = [&](size_t bytes) -> char* {
    char* p = (char*)d_ws + off;
    off += (bytes + 255) & ~(size_t)255;
    return p;
  };

  __bf16* enc_wih_bf  = (__bf16*)alloc((size_t)Gn * Hn * 2);
  __bf16* enc_whh_bf  = (__bf16*)alloc((size_t)Gn * Hn * 2);
  __bf16* dec_wih_bf  = (__bf16*)alloc((size_t)Gn * Hn * 2);
  __bf16* dec_whh_bf  = (__bf16*)alloc((size_t)Gn * Hn * 2);
  __bf16* glm_wq_bf   = (__bf16*)alloc((size_t)Hn * Hn * 2);
  __bf16* glm_wref_bf = (__bf16*)alloc((size_t)Hn * Hn * 2);
  __bf16* ptr_wq_bf   = (__bf16*)alloc((size_t)Hn * Hn * 2);
  __bf16* ptr_wref_bf = (__bf16*)alloc((size_t)Hn * Hn * 2);
  __bf16* emb_bf      = (__bf16*)alloc((size_t)Bn * Sn * Hn * 2);
  __bf16* enc_out_bf  = (__bf16*)alloc((size_t)Bn * Sn * Hn * 2);
  __bf16* ref_g_bf    = (__bf16*)alloc((size_t)Bn * Sn * Hn * 2);
  __bf16* ref_p_bf    = (__bf16*)alloc((size_t)Bn * Sn * Hn * 2);
  float*  gates       = (float*) alloc((size_t)Bn * Gn * 4);
  float*  h           = (float*) alloc((size_t)Bn * Hn * 4);
  float*  c           = (float*) alloc((size_t)Bn * Hn * 4);
  __bf16* h_bf        = (__bf16*)alloc((size_t)Bn * Hn * 2);
  float*  bias_enc    = (float*) alloc((size_t)Gn * 4);
  float*  bias_dec    = (float*) alloc((size_t)Gn * 4);
  float*  qproj       = (float*) alloc((size_t)Bn * Hn * 4);
  float*  probs       = (float*) alloc((size_t)Bn * Sn * 4);
  __bf16* q_bf        = (__bf16*)alloc((size_t)Bn * Hn * 2);
  __bf16* dec_in_bf   = (__bf16*)alloc((size_t)Bn * Hn * 2);
  int*    mask        = (int*)   alloc((size_t)Bn * Sn * 4);
  int*    chosen_i    = (int*)   alloc((size_t)Bn * 4);

  float* out_lp = (float*)d_out;
  float* out_ch = out_lp + (size_t)Bn * Sn;

  const dim3 blk(256);
  auto nb = [](int n) { return dim3((n + 255) / 256); };

  // ---- weight conversion + bias fusion ----
  ptrnet_cvt_f2bf<<<nb(Gn * Hn), blk, 0, stream>>>(enc_wih, enc_wih_bf, Gn * Hn);
  ptrnet_cvt_f2bf<<<nb(Gn * Hn), blk, 0, stream>>>(enc_whh, enc_whh_bf, Gn * Hn);
  ptrnet_cvt_f2bf<<<nb(Gn * Hn), blk, 0, stream>>>(dec_wih, dec_wih_bf, Gn * Hn);
  ptrnet_cvt_f2bf<<<nb(Gn * Hn), blk, 0, stream>>>(dec_whh, dec_whh_bf, Gn * Hn);
  ptrnet_cvt_f2bf<<<nb(Hn * Hn), blk, 0, stream>>>(glm_wq_w,   glm_wq_bf,   Hn * Hn);
  ptrnet_cvt_f2bf<<<nb(Hn * Hn), blk, 0, stream>>>(glm_wref_w, glm_wref_bf, Hn * Hn);
  ptrnet_cvt_f2bf<<<nb(Hn * Hn), blk, 0, stream>>>(ptr_wq_w,   ptr_wq_bf,   Hn * Hn);
  ptrnet_cvt_f2bf<<<nb(Hn * Hn), blk, 0, stream>>>(ptr_wref_w, ptr_wref_bf, Hn * Hn);
  ptrnet_addvec<<<nb(Gn), blk, 0, stream>>>(enc_bih, enc_bhh, bias_enc, Gn);
  ptrnet_addvec<<<nb(Gn), blk, 0, stream>>>(dec_bih, dec_bhh, bias_dec, Gn);

  // ---- state init ----
  ptrnet_fillf <<<nb(Bn * Hn), blk, 0, stream>>>(h, Bn * Hn);
  ptrnet_fillf <<<nb(Bn * Hn), blk, 0, stream>>>(c, Bn * Hn);
  ptrnet_fillbf<<<nb(Bn * Hn), blk, 0, stream>>>(h_bf, Bn * Hn);
  ptrnet_filli <<<nb(Bn * Sn), blk, 0, stream>>>(mask, Bn * Sn);

  // ---- embedding ----
  ptrnet_embed<<<nb(Bn * Sn * Hn), blk, 0, stream>>>(inputs, emb_w, emb_b, emb_bf);

  const int gridGates = (Bn / TM) * (Gn / TN);        // 512
  const int gridQ     = (Bn / TM) * (Hn / TN);        // 128
  const int gridRef   = ((Bn * Sn) / TM) * (Hn / TN); // 12800

  // ---- encoder scan ----
  for (int t = 0; t < Sn; ++t) {
    ptrnet_gemm_bf16_wmma<<<gridGates, blk, 0, stream>>>(
        emb_bf + (size_t)t * Hn, Sn * Hn, enc_wih_bf,
        h_bf, Hn, enc_whh_bf,
        bias_enc, gates, nullptr, Bn, Gn, Hn, Hn);
    ptrnet_lstm_cell<<<nb(Bn * Hn), blk, 0, stream>>>(
        gates, h, c, h_bf, enc_out_bf + (size_t)t * Hn);
  }

  // ---- ref projections (bf16 output) ----
  ptrnet_gemm_bf16_wmma<<<gridRef, blk, 0, stream>>>(
      enc_out_bf, Hn, glm_wref_bf, nullptr, 0, nullptr,
      glm_wref_b, nullptr, ref_g_bf, Bn * Sn, Hn, Hn, 0);
  ptrnet_gemm_bf16_wmma<<<gridRef, blk, 0, stream>>>(
      enc_out_bf, Hn, ptr_wref_bf, nullptr, 0, nullptr,
      ptr_wref_b, nullptr, ref_p_bf, Bn * Sn, Hn, Hn, 0);

  // ---- decoder ----
  ptrnet_start_bcast<<<nb(Bn * Hn), blk, 0, stream>>>(startv, dec_in_bf);

  for (int st = 0; st < Sn; ++st) {
    ptrnet_gemm_bf16_wmma<<<gridGates, blk, 0, stream>>>(
        dec_in_bf, Hn, dec_wih_bf,
        h_bf, Hn, dec_whh_bf,
        bias_dec, gates, nullptr, Bn, Gn, Hn, Hn);
    ptrnet_lstm_cell<<<nb(Bn * Hn), blk, 0, stream>>>(gates, h, c, h_bf, nullptr);

    // glimpse attention: qproj = h @ glm_wq^T + b
    ptrnet_gemm_bf16_wmma<<<gridQ, blk, 0, stream>>>(
        h_bf, Hn, glm_wq_bf, nullptr, 0, nullptr,
        glm_wq_b, qproj, nullptr, Bn, Hn, Hn, 0);
    ptrnet_attn<<<Bn, blk, 0, stream>>>(
        ref_g_bf, qproj, glm_v_w, glm_v_b, mask, probs,
        nullptr, nullptr, nullptr, st, 0);
    ptrnet_glimpse<<<nb(Bn * Hn), blk, 0, stream>>>(probs, enc_out_bf, q_bf);

    // pointer attention: qproj = q @ ptr_wq^T + b
    ptrnet_gemm_bf16_wmma<<<gridQ, blk, 0, stream>>>(
        q_bf, Hn, ptr_wq_bf, nullptr, 0, nullptr,
        ptr_wq_b, qproj, nullptr, Bn, Hn, Hn, 0);
    ptrnet_attn<<<Bn, blk, 0, stream>>>(
        ref_p_bf, qproj, ptr_v_w, ptr_v_b, mask, nullptr,
        out_lp, out_ch, chosen_i, st, 1);

    ptrnet_gather<<<Bn, blk, 0, stream>>>(emb_bf, chosen_i, dec_in_bf);
  }
}
